// SubGCon2_32074815766916
// MI455X (gfx1250) — matrix operand order repaired and checked by reading, not compiled, not run on hardware
//
#include <hip/hip_runtime.h>
#include <hip/hip_bf16.h>

#define NA 40000
#define NP 100000
#define NT 20000
#define DIM 128
#define OUTD 64
#define EAP 500000
#define EPT 800000
#define RPB 32          // rows per block in sage_gemm (2 row tiles of 16)
#define LDA 136         // LDS row stride in bf16 (272B = 17*16B: aligned, conflict-free)

static_assert(NA % RPB == 0 && NP % RPB == 0 && NT % RPB == 0, "row blocks exact");

typedef __attribute__((ext_vector_type(16))) __bf16 v16bf;
typedef __attribute__((ext_vector_type(8)))  float  v8f;

// ---------------------------------------------------------------------------
// B 32x16 bf16 fragment from pre-transposed WT[n][k] (row-major, 128 wide):
// lane L -> col N=L%16; elems 0..7 at K=kb+8*(L/16), elems 8..15 at +16.
// ---------------------------------------------------------------------------
__device__ __forceinline__ v16bf load_b(const __bf16* __restrict__ WT, int col,
                                        int kb, int lane) {
  const __bf16* p = WT + (size_t)col * DIM + kb + ((lane >> 4) << 3);
  v16bf b;
#pragma unroll
  for (int j = 0; j < 8; ++j) b[j] = p[j];
#pragma unroll
  for (int j = 0; j < 8; ++j) b[j + 8] = p[j + 16];
  return b;
}

// A 16x32 bf16 fragment from LDS tile (row-major, stride LDA):
// lane L -> row M=L%16; elems 0..7 at k=kb+8*(L/16), elems 8..15 at +16.
// Both 16B chunks are 16B-aligned -> ds_load_b128 pairs.
__device__ __forceinline__ v16bf lds_a(const __bf16* Lbase, int row, int kb,
                                       int lane) {
  const __bf16* p = Lbase + row * LDA + kb + ((lane >> 4) << 3);
  v16bf a;
#pragma unroll
  for (int j = 0; j < 8; ++j) a[j] = p[j];
#pragma unroll
  for (int j = 0; j < 8; ++j) a[j + 8] = p[j + 16];
  return a;
}

// Convert 16 contiguous fp32 (scaled) -> 16 contiguous bf16 in LDS (32B, aligned).
__device__ __forceinline__ void stage16(__bf16* __restrict__ dst,
                                        const float* __restrict__ src, float s) {
#pragma unroll
  for (int j = 0; j < 16; j += 4) {
    float4 v = *(const float4*)(src + j);
    dst[j + 0] = (__bf16)(v.x * s);
    dst[j + 1] = (__bf16)(v.y * s);
    dst[j + 2] = (__bf16)(v.z * s);
    dst[j + 3] = (__bf16)(v.w * s);
  }
}

// ---------------------------------------------------------------------------
// Fused SAGE update: out = mean1@Wl1 + [mean2@Wl2] + xdst@Wr + bias, relu.
// Block = 256 (8 waves); wave w owns output-column tile w; block owns 32 rows.
// A tiles staged once to LDS (bf16); B fragments hoisted into registers.
// ---------------------------------------------------------------------------
__global__ __launch_bounds__(256)
void sage_gemm(const float* __restrict__ sum1, const float* __restrict__ cnt1,
               const __bf16* __restrict__ Wl1T,
               const float* __restrict__ sum2, const float* __restrict__ cnt2,
               const __bf16* __restrict__ Wl2T,
               const float* __restrict__ xdst, const __bf16* __restrict__ WrT,
               const float* __restrict__ bias, float* __restrict__ out,
               int has2) {
  __shared__ alignas(16) __bf16 La[3][RPB][LDA];

  const int tid  = threadIdx.x;
  const int wave = tid >> 5;
  const int lane = tid & 31;
  const int row0 = blockIdx.x * RPB;

  // ---- stage A tiles (xdst, mean1, [mean2]) into LDS, converted to bf16 ----
  {
    const int r  = tid >> 3;            // 0..31: local row
    const int c0 = (tid & 7) << 4;      // 0,16,...,112
    const int g  = row0 + r;
    stage16(&La[0][r][c0], xdst + (size_t)g * DIM + c0, 1.0f);
    const float rcp1 = 1.0f / fmaxf(cnt1[g], 1.0f);
    stage16(&La[1][r][c0], sum1 + (size_t)g * DIM + c0, rcp1);
    if (has2) {
      const float rcp2 = 1.0f / fmaxf(cnt2[g], 1.0f);
      stage16(&La[2][r][c0], sum2 + (size_t)g * DIM + c0, rcp2);
    }
  }

  // ---- hoist all B fragments for this wave's column tile (loop-invariant) ----
  const int ncol = wave * 16 + (lane & 15);
  v16bf BR[4], B1[4], B2[4];
#pragma unroll
  for (int kbi = 0; kbi < 4; ++kbi) {
    BR[kbi] = load_b(WrT, ncol, kbi * 32, lane);
    B1[kbi] = load_b(Wl1T, ncol, kbi * 32, lane);
    if (has2) B2[kbi] = load_b(Wl2T, ncol, kbi * 32, lane);
  }

  __syncthreads();

  const float bb = bias[ncol];
#pragma unroll
  for (int rt = 0; rt < 2; ++rt) {
    const int rb   = rt * 16;
    const int arow = rb + (lane & 15);
    v8f acc = {};
#pragma unroll
    for (int kbi = 0; kbi < 4; ++kbi) {
      const int kb = kbi * 32;
      v16bf aX = lds_a(&La[0][0][0], arow, kb, lane);
      acc = __builtin_amdgcn_wmma_f32_16x16x32_bf16(false, aX, false, BR[kbi],
                                                    (short)0, acc, false, false);
      v16bf a1 = lds_a(&La[1][0][0], arow, kb, lane);
      acc = __builtin_amdgcn_wmma_f32_16x16x32_bf16(false, a1, false, B1[kbi],
                                                    (short)0, acc, false, false);
      if (has2) {  // uniform branch: EXEC stays all-ones
        v16bf a2 = lds_a(&La[2][0][0], arow, kb, lane);
        acc = __builtin_amdgcn_wmma_f32_16x16x32_bf16(false, a2, false, B2[kbi],
                                                      (short)0, acc, false, false);
      }
    }
    // C/D layout: VGPR r -> M = r + 8*(lane/16), N = lane%16
    const int mbase = row0 + rb + ((lane >> 4) << 3);
#pragma unroll
    for (int r = 0; r < 8; ++r) {
      float v = acc[r] + bb;
      out[(size_t)(mbase + r) * DIM + ncol] = fmaxf(v, 0.0f);
    }
  }
}

// ---------------------------------------------------------------------------
// Head GEMM: out[n x 64] = x[n x 128] @ WT(64x128)^T + bias, no relu.
// Small share of total work; direct-load variant.
// ---------------------------------------------------------------------------
__device__ __forceinline__ v16bf load_a_g(const float* __restrict__ base,
                                          int row, int kb, int lane) {
  const float* p = base + (size_t)row * DIM + kb + ((lane >> 4) << 3);
  v16bf a;
#pragma unroll
  for (int j = 0; j < 8; ++j) a[j] = (__bf16)p[j];
#pragma unroll
  for (int j = 0; j < 8; ++j) a[j + 8] = (__bf16)p[j + 16];
  return a;
}

__global__ __launch_bounds__(128)
void head_gemm(const float* __restrict__ x, const __bf16* __restrict__ WT,
               const float* __restrict__ bias, float* __restrict__ out) {
  const int wave = threadIdx.x >> 5;  // 0..3 -> 64 cols
  const int lane = threadIdx.x & 31;
  const int row0 = blockIdx.x * 16;
  const int arow = row0 + (lane & 15);
  const int ncol = wave * 16 + (lane & 15);
  v8f acc = {};
#pragma unroll
  for (int kb = 0; kb < DIM; kb += 32) {
    v16bf a = load_a_g(x, arow, kb, lane);
    v16bf b = load_b(WT, ncol, kb, lane);
    acc = __builtin_amdgcn_wmma_f32_16x16x32_bf16(false, a, false, b,
                                                  (short)0, acc, false, false);
  }
  const float bb = bias[ncol];
  const int mbase = row0 + ((lane >> 4) << 3);
#pragma unroll
  for (int r = 0; r < 8; ++r)
    out[(size_t)(mbase + r) * OUTD + ncol] = acc[r] + bb;
}

// ---------------------------------------------------------------------------
// Edge aggregation: one wave per edge, float4 per lane (fully coalesced),
// native f32 global atomics.
// ---------------------------------------------------------------------------
__global__ __launch_bounds__(256)
void scatter_add(const float* __restrict__ x, const int* __restrict__ src,
                 const int* __restrict__ dst, float* __restrict__ sum, int nE) {
  int t = blockIdx.x * blockDim.x + threadIdx.x;
  int e = t >> 5;
  if (e >= nE) return;
  int lane = t & 31;
  int s = src[e], d = dst[e];
  const float4 v = ((const float4*)(x + (size_t)s * DIM))[lane];
  float* o = sum + (size_t)d * DIM + lane * 4;
  unsafeAtomicAdd(o + 0, v.x);
  unsafeAtomicAdd(o + 1, v.y);
  unsafeAtomicAdd(o + 2, v.z);
  unsafeAtomicAdd(o + 3, v.w);
}

__global__ __launch_bounds__(256)
void scatter_cnt(const int* __restrict__ dst, float* __restrict__ cnt, int nE) {
  int e = blockIdx.x * blockDim.x + threadIdx.x;
  if (e < nE) unsafeAtomicAdd(&cnt[dst[e]], 1.0f);
}

// ---------------------------------------------------------------------------
// Weight prep: transpose 128x128 fp32 (W[k][n]) -> bf16 WT[n][k].
// ---------------------------------------------------------------------------
__global__ void prep_T(const float* __restrict__ W, __bf16* __restrict__ WT) {
  int n = blockIdx.x, k = threadIdx.x;
  WT[(size_t)n * DIM + k] = (__bf16)W[(size_t)k * DIM + n];
}
__global__ void prep_T_sum(const float* __restrict__ Wa,
                           const float* __restrict__ Wb,
                           __bf16* __restrict__ WT) {
  int n = blockIdx.x, k = threadIdx.x;
  WT[(size_t)n * DIM + k] =
      (__bf16)(Wa[(size_t)k * DIM + n] + Wb[(size_t)k * DIM + n]);
}
__global__ void prep_T_head(const float* __restrict__ W,  // [128][64]
                            __bf16* __restrict__ WT) {    // [64][128]
  int n = blockIdx.x, k = threadIdx.x;
  WT[(size_t)n * DIM + k] = (__bf16)W[(size_t)k * OUTD + n];
}
__global__ void add_bias128(const float* __restrict__ a,
                            const float* __restrict__ b,
                            float* __restrict__ o) {
  o[threadIdx.x] = a[threadIdx.x] + b[threadIdx.x];
}

// temperature = ll1 @ w2 + b2 ; out = logits / temperature
__global__ __launch_bounds__(256)
void temp_div(const float* __restrict__ logits, const float* __restrict__ ll1,
              const float* __restrict__ w2, const float* __restrict__ b2,
              float* __restrict__ out, int n) {
  int i = blockIdx.x * blockDim.x + threadIdx.x;
  if (i >= n) return;
  float t = b2[0];
#pragma unroll
  for (int j = 0; j < OUTD; ++j) t += ll1[(size_t)i * OUTD + j] * w2[j];
  float rt = 1.0f / t;
#pragma unroll
  for (int j = 0; j < OUTD; ++j)
    out[(size_t)i * OUTD + j] = logits[(size_t)i * OUTD + j] * rt;
}

// ---------------------------------------------------------------------------
extern "C" void kernel_launch(void* const* d_in, const int* in_sizes, int n_in,
                              void* d_out, int out_size, void* d_ws,
                              size_t ws_size, hipStream_t stream) {
  const float* x_author = (const float*)d_in[0];
  const float* x_paper  = (const float*)d_in[1];
  const float* x_term   = (const float*)d_in[2];
  const float* W_l      = (const float*)d_in[3];   // [3,2,4,128,128]
  const float* b_l      = (const float*)d_in[4];   // [3,2,4,128]
  const float* W_r      = (const float*)d_in[5];
  const float* lmW      = (const float*)d_in[6];
  const float* lmB      = (const float*)d_in[7];
  const float* ltW      = (const float*)d_in[8];
  const float* ltB      = (const float*)d_in[9];
  const float* l2W      = (const float*)d_in[10];
  const float* l2B      = (const float*)d_in[11];
  const int* src_ap = (const int*)d_in[12];
  const int* dst_ap = (const int*)d_in[13];
  const int* src_pa = (const int*)d_in[14];
  const int* dst_pa = (const int*)d_in[15];
  const int* src_pt = (const int*)d_in[16];
  const int* dst_pt = (const int*)d_in[17];
  const int* src_tp = (const int*)d_in[18];
  const int* dst_tp = (const int*)d_in[19];

  // ---- workspace carve ----
  char* w = (char*)d_ws;
  auto alloc = [&](size_t bytes) -> char* {
    char* p = w;
    w += (bytes + 255) & ~(size_t)255;
    return p;
  };
  float* sum_ap = (float*)alloc((size_t)NP * DIM * 4);
  float* sum_tp = (float*)alloc((size_t)NP * DIM * 4);
  float* sum_pa = (float*)alloc((size_t)NA * DIM * 4);
  float* sum_pt = (float*)alloc((size_t)NT * DIM * 4);
  float* cnt_ap = (float*)alloc((size_t)NP * 4);
  float* cnt_tp = (float*)alloc((size_t)NP * 4);
  float* cnt_pa = (float*)alloc((size_t)NA * 4);
  float* cnt_pt = (float*)alloc((size_t)NT * 4);
  float* xa[2] = {(float*)alloc((size_t)NA * DIM * 4), (float*)alloc((size_t)NA * DIM * 4)};
  float* xp[2] = {(float*)alloc((size_t)NP * DIM * 4), (float*)alloc((size_t)NP * DIM * 4)};
  float* xt[2] = {(float*)alloc((size_t)NT * DIM * 4), (float*)alloc((size_t)NT * DIM * 4)};
  float* logits = (float*)alloc((size_t)NA * OUTD * 4);
  float* ll1    = (float*)alloc((size_t)NA * OUTD * 4);
  __bf16* WlT   = (__bf16*)alloc((size_t)24 * DIM * DIM * 2);  // per (sl,rel)
  __bf16* WrT1  = (__bf16*)alloc((size_t)6 * DIM * DIM * 2);   // rel p->a
  __bf16* WrT2  = (__bf16*)alloc((size_t)6 * DIM * DIM * 2);   // rel p->t
  __bf16* WrTp  = (__bf16*)alloc((size_t)6 * DIM * DIM * 2);   // combined rel0+rel3
  __bf16* WTm   = (__bf16*)alloc((size_t)OUTD * DIM * 2);
  __bf16* WTt   = (__bf16*)alloc((size_t)OUTD * DIM * 2);
  float* biasP  = (float*)alloc((size_t)6 * DIM * 4);

  const size_t MSZ = (size_t)DIM * DIM;  // 16384 elements

  // ---- weight prep (once per call; deterministic) ----
  for (int sl = 0; sl < 6; ++sl) {
    for (int rel = 0; rel < 4; ++rel)
      prep_T<<<DIM, DIM, 0, stream>>>(W_l + (size_t)(sl * 4 + rel) * MSZ,
                                      WlT + (size_t)(sl * 4 + rel) * MSZ);
    prep_T<<<DIM, DIM, 0, stream>>>(W_r + (size_t)(sl * 4 + 1) * MSZ, WrT1 + (size_t)sl * MSZ);
    prep_T<<<DIM, DIM, 0, stream>>>(W_r + (size_t)(sl * 4 + 2) * MSZ, WrT2 + (size_t)sl * MSZ);
    prep_T_sum<<<DIM, DIM, 0, stream>>>(W_r + (size_t)(sl * 4 + 0) * MSZ,
                                        W_r + (size_t)(sl * 4 + 3) * MSZ,
                                        WrTp + (size_t)sl * MSZ);
    add_bias128<<<1, DIM, 0, stream>>>(b_l + (size_t)(sl * 4 + 0) * DIM,
                                       b_l + (size_t)(sl * 4 + 3) * DIM,
                                       biasP + (size_t)sl * DIM);
  }
  prep_T_head<<<OUTD, DIM, 0, stream>>>(lmW, WTm);
  prep_T_head<<<OUTD, DIM, 0, stream>>>(ltW, WTt);

  // ---- degree counts (edge structure is layer-invariant) ----
  hipMemsetAsync(cnt_ap, 0, (size_t)NP * 4, stream);
  hipMemsetAsync(cnt_tp, 0, (size_t)NP * 4, stream);
  hipMemsetAsync(cnt_pa, 0, (size_t)NA * 4, stream);
  hipMemsetAsync(cnt_pt, 0, (size_t)NT * 4, stream);
  scatter_cnt<<<(EAP + 255) / 256, 256, 0, stream>>>(dst_ap, cnt_ap, EAP);
  scatter_cnt<<<(EAP + 255) / 256, 256, 0, stream>>>(dst_pa, cnt_pa, EAP);
  scatter_cnt<<<(EPT + 255) / 256, 256, 0, stream>>>(dst_pt, cnt_pt, EPT);
  scatter_cnt<<<(EPT + 255) / 256, 256, 0, stream>>>(dst_tp, cnt_tp, EPT);

  auto egrid = [](int nE) { return ((size_t)nE * 32 + 255) / 256; };

  auto run_layer = [&](int sl, const float* xa_in, const float* xp_in,
                       const float* xt_in, float* xa_out, float* xp_out,
                       float* xt_out, bool do_p, bool do_t) {
    if (do_p) {
      hipMemsetAsync(sum_ap, 0, (size_t)NP * DIM * 4, stream);
      hipMemsetAsync(sum_tp, 0, (size_t)NP * DIM * 4, stream);
      scatter_add<<<egrid(EAP), 256, 0, stream>>>(xa_in, src_ap, dst_ap, sum_ap, EAP);
      scatter_add<<<egrid(EPT), 256, 0, stream>>>(xt_in, src_tp, dst_tp, sum_tp, EPT);
    }
    hipMemsetAsync(sum_pa, 0, (size_t)NA * DIM * 4, stream);
    scatter_add<<<egrid(EAP), 256, 0, stream>>>(xp_in, src_pa, dst_pa, sum_pa, EAP);
    if (do_t) {
      hipMemsetAsync(sum_pt, 0, (size_t)NT * DIM * 4, stream);
      scatter_add<<<egrid(EPT), 256, 0, stream>>>(xp_in, src_pt, dst_pt, sum_pt, EPT);
    }
    if (do_p)
      sage_gemm<<<NP / RPB, 256, 0, stream>>>(
          sum_ap, cnt_ap, WlT + (size_t)(sl * 4 + 0) * MSZ,
          sum_tp, cnt_tp, WlT + (size_t)(sl * 4 + 3) * MSZ,
          xp_in, WrTp + (size_t)sl * MSZ, biasP + (size_t)sl * DIM, xp_out, 1);
    sage_gemm<<<NA / RPB, 256, 0, stream>>>(
        sum_pa, cnt_pa, WlT + (size_t)(sl * 4 + 1) * MSZ,
        nullptr, nullptr, nullptr,
        xa_in, WrT1 + (size_t)sl * MSZ, b_l + (size_t)(sl * 4 + 1) * DIM, xa_out, 0);
    if (do_t)
      sage_gemm<<<NT / RPB, 256, 0, stream>>>(
          sum_pt, cnt_pt, WlT + (size_t)(sl * 4 + 2) * MSZ,
          nullptr, nullptr, nullptr,
          xt_in, WrT2 + (size_t)sl * MSZ, b_l + (size_t)(sl * 4 + 2) * DIM, xt_out, 0);
  };

  float* out0 = (float*)d_out;                    // logits / temperature
  float* oa = out0 + (size_t)NA * OUTD;           // stack-2 author features
  float* op = oa + (size_t)NA * DIM;
  float* ot = op + (size_t)NP * DIM;

  // stack 0 (sl 0,1): raw features -> (ma, mp, mt)
  run_layer(0, x_author, x_paper, x_term, xa[0], xp[0], xt[0], true, true);
  run_layer(1, xa[0], xp[0], xt[0], xa[1], xp[1], xt[1], true, true);
  head_gemm<<<NA / 16, 128, 0, stream>>>(xa[1], WTm, lmB, logits);

  // stack 1 (sl 2,3): second layer only needs the author update (-> ta)
  run_layer(2, xa[1], xp[1], xt[1], xa[0], xp[0], xt[0], true, true);
  run_layer(3, xa[0], xp[0], xt[0], xa[1], xp[1], xt[1], false, false);
  head_gemm<<<NA / 16, 128, 0, stream>>>(xa[1], WTt, ltB, ll1);
  temp_div<<<(NA + 255) / 256, 256, 0, stream>>>(logits, ll1, l2W, l2B, out0, NA);

  // stack 2 (sl 4,5): raw features -> outputs (post-relu), written to d_out
  run_layer(4, x_author, x_paper, x_term, xa[0], xp[0], xt[0], true, true);
  run_layer(5, xa[0], xp[0], xt[0], oa, op, ot, true, true);
}
